// KPConvInterSO3_77146202570856
// MI455X (gfx1250) — matrix operand order repaired and checked by reading, compiled, not run on hardware
//
#include <hip/hip_runtime.h>
#include <math.h>

// KPConv / E2PN InterSO3 fused kernel for gfx1250 (MI455X), wave32 + WMMA.
// N=16000 queries, M=16000 support, NB=32 neighbors, A=4 anchors,
// K=15 kernel points (padded to 16), CIN=COUT=32.

#define QN   16000
#define NBR  32
#define AK   4
#define KPN  15
#define CIN_ 32
#define COUT_ 32
#define KP_EXT_INV 1.6666667f   // 1/0.6

typedef __attribute__((ext_vector_type(16))) _Float16 v16h;
typedef __attribute__((ext_vector_type(2)))  _Float16 v2h;
typedef __attribute__((ext_vector_type(8)))  float    v8f;

union H16 { v16h v; uint4 q[2]; };   // 16 halves <-> 2x b128

// Dynamic LDS layout (bytes):
//   [0,      8192)  float4 nv[16*32]    neighbor vecs (dx,dy,dz,|d|^2), anchor-independent
//   [8192,   9216)  float4 rkp[4*16]    rotated kernel pts (x,y,z,|kp|^2); k=15 sentinel k2=1e8
//   [9216,  11264)  int    nbl[16*32]   neighbor indices for this tile
//   [11264, 44032)  f16    w2t[32*512]  W2 transposed: [d][k*32+c], k=15 zero-padded
//   [44032,109568)  f16    wg[4*16*512] per-(anchor,n): KP weights [k][b], then reused as G [kc]
#define SMEM_BYTES 109568

__global__ __launch_bounds__(128)
void kpconv_e2pn_fused(const float* __restrict__ q_pts,
                       const float* __restrict__ s_pts,
                       const int*   __restrict__ nbi,
                       const float* __restrict__ x,       // [M, A, CIN]
                       const float* __restrict__ kp,      // [15, 3]
                       const float* __restrict__ anc,     // [4, 3, 3]
                       const float* __restrict__ wts,     // [15, CIN, COUT]
                       float*       __restrict__ out)     // [N, A, COUT]
{
    extern __shared__ char smem[];
    float4*   nv  = (float4*)(smem);
    float4*   rkp = (float4*)(smem + 8192);
    int*      nbl = (int*)   (smem + 9216);
    _Float16* w2t = (_Float16*)(smem + 11264);
    _Float16* wg  = (_Float16*)(smem + 44032);

    const int tid  = threadIdx.x;
    const int lane = tid & 31;
    const int wave = tid >> 5;      // = anchor index
    const int tile = blockIdx.x;    // 16 queries per block
    const int hl   = lane >> 4;     // wave half (0/1) -- selects K-range per ISA layouts
    const int lm   = lane & 15;     // column / row index within 16

    // ---------------- Phase 0: cooperative staging ----------------
    for (int i = tid; i < 16 * NBR; i += 128) {
        const int n = i >> 5, b = i & 31;
        const int m = nbi[(tile * 16 + n) * NBR + b];
        nbl[i] = m;
        const float qx = q_pts[(tile * 16 + n) * 3 + 0];
        const float qy = q_pts[(tile * 16 + n) * 3 + 1];
        const float qz = q_pts[(tile * 16 + n) * 3 + 2];
        const float dx = s_pts[m * 3 + 0] - qx;
        const float dy = s_pts[m * 3 + 1] - qy;
        const float dz = s_pts[m * 3 + 2] - qz;
        nv[i] = make_float4(dx, dy, dz, dx * dx + dy * dy + dz * dz);
    }
    // Rotated kernel points per anchor; padded k=15 gets huge |kp|^2 so its
    // KP influence clamps to exactly 0.
    if (tid < 64) {
        const int a = tid >> 4, k = tid & 15;
        if (k < KPN) {
            const float px = kp[k * 3 + 0], py = kp[k * 3 + 1], pz = kp[k * 3 + 2];
            const float* R = anc + a * 9;
            const float rx = R[0] * px + R[1] * py + R[2] * pz;
            const float ry = R[3] * px + R[4] * py + R[5] * pz;
            const float rz = R[6] * px + R[7] * py + R[8] * pz;
            rkp[tid] = make_float4(rx, ry, rz, rx * rx + ry * ry + rz * rz);
        } else {
            rkp[tid] = make_float4(0.f, 0.f, 0.f, 1.0e8f);
        }
    }
    // W2^T staging with packed b32 stores: w2t[d][kc], kc = k*32 + c.
    for (int i = tid; i < 32 * 256; i += 128) {
        const int d   = i >> 8;
        const int kc0 = (i & 255) * 2;
        const int kc1 = kc0 + 1;
        const int k0 = kc0 >> 5, c0 = kc0 & 31;
        const int k1 = kc1 >> 5, c1 = kc1 & 31;
        const float f0 = (k0 < KPN) ? wts[(k0 * CIN_ + c0) * COUT_ + d] : 0.f;
        const float f1 = (k1 < KPN) ? wts[(k1 * CIN_ + c1) * COUT_ + d] : 0.f;
        v2h p; p[0] = (_Float16)f0; p[1] = (_Float16)f1;
        *(v2h*)(w2t + d * 512 + kc0) = p;
    }
    __syncthreads();

    const int a = wave;   // this wave's anchor

    // ---------------- Phase 1: KP influence weights ----------------
    // Lane role: kernel point k = lm, b-half = hl. One rkp per lane (register),
    // nv reads are LDS broadcasts, stores are 2x ds_store_b128 per query.
    {
        const float4 r = rkp[a * 16 + lm];
        for (int n = 0; n < 16; ++n) {
            H16 u;
            #pragma unroll
            for (int e = 0; e < 16; ++e) {
                const float4 v = nv[n * NBR + hl * 16 + e];
                const float cr = v.x * r.x + v.y * r.y + v.z * r.z;
                const float sq = fmaxf(v.w + r.w - 2.0f * cr, 1e-12f);
                u.v[e] = (_Float16)fmaxf(0.0f, 1.0f - sqrtf(sq) * KP_EXT_INV);
            }
            uint4* dst = (uint4*)(wg + (a * 16 + n) * 512 + lm * NBR + hl * 16);
            dst[0] = u.q[0];
            dst[1] = u.q[1];
        }
    }
    // No barrier: wg[a] is wave-private and DS ops are in-order per wave.

    // ---------------- Phase 2: GEMM1 per query ----------------
    // G[k,c] = sum_b w[b,k] * x[nbr(b), a, c]   via v_wmma_f32_16x16x32_f16.
    // A (16x32, rows=k, K=b): lanes 0-15 hold b {0..15}; lanes 16-31 {16..31}.
    // B (32x16, K=b, cols): column permutation tile0 j -> c=2j, tile1 j -> c=2j+1,
    // so each lane gathers one b64 (adjacent c pair) per element and the G pair
    // lands at adjacent kc addresses (one packed b32 store).
    for (int n = 0; n < 16; ++n) {
        const _Float16* wrow = wg + (a * 16 + n) * 512;   // [k*32 + b]
        const int b0 = hl * 16;
        const int b1 = hl * 16 + 8;
        H16 ua;
        ua.q[0] = *(const uint4*)(wrow + lm * NBR + b0);
        ua.q[1] = *(const uint4*)(wrow + lm * NBR + b1);

        const int* nbn = nbl + n * NBR;
        const int  bb  = hl * 16;
        v16h B0, B1;
        #pragma unroll
        for (int e = 0; e < 16; ++e) {
            const int m = nbn[bb + e];
            const float2 f = *(const float2*)(x + ((size_t)m * AK + a) * CIN_ + 2 * lm);
            B0[e] = (_Float16)f.x;
            B1[e] = (_Float16)f.y;
        }

        v8f z = {};
        v8f g0 = __builtin_amdgcn_wmma_f32_16x16x32_f16(false, ua.v, false, B0,
                                                        (short)0, z, false, false);
        v8f g1 = __builtin_amdgcn_wmma_f32_16x16x32_f16(false, ua.v, false, B1,
                                                        (short)0, z, false, false);

        // Stage G as GEMM2 A rows: A2[n][kc], kc = k*32 + c; (g0[r],g1[r]) are
        // (c=2*lm, c=2*lm+1) -> one packed b32 store each.
        // C/D layout: lane%16 = column, VGPR r = row r + 8*(lane/16).
        _Float16* grow = wg + (a * 16 + n) * 512;   // aliases wrow (reads already done)
        #pragma unroll
        for (int r = 0; r < 8; ++r) {
            const int M = r + hl * 8;
            v2h p; p[0] = (_Float16)g0[r]; p[1] = (_Float16)g1[r];
            *(v2h*)(grow + M * 32 + 2 * lm) = p;
        }
    }

    // ---------------- Phase 3: GEMM2 (16 queries batched as rows) ----------------
    // out[n,a,d] = sum_{kc} G[n][kc] * W2[kc][d]; 16 K-steps of 32, 2 column tiles.
    v8f acc0 = {}, acc1 = {};
    for (int kk = 0; kk < 16; ++kk) {
        const _Float16* arow = wg + (a * 16 + lm) * 512 + kk * 32;  // row n = lane%16
        H16 ua;
        ua.q[0] = *(const uint4*)(arow + hl * 16);
        ua.q[1] = *(const uint4*)(arow + hl * 16 + 8);

        const int kb = kk * 32 + hl * 16;
        const _Float16* w0 = w2t + (size_t)lm * 512 + kb;          // col d = lm
        const _Float16* w1 = w2t + (size_t)(lm + 16) * 512 + kb;   // col d = lm+16
        H16 ub0, ub1;
        ub0.q[0] = *(const uint4*)(w0);
        ub0.q[1] = *(const uint4*)(w0 + 8);
        ub1.q[0] = *(const uint4*)(w1);
        ub1.q[1] = *(const uint4*)(w1 + 8);

        acc0 = __builtin_amdgcn_wmma_f32_16x16x32_f16(false, ua.v, false, ub0.v,
                                                      (short)0, acc0, false, false);
        acc1 = __builtin_amdgcn_wmma_f32_16x16x32_f16(false, ua.v, false, ub1.v,
                                                      (short)0, acc1, false, false);
    }

    // Store out[n, a, d] (f32).
    #pragma unroll
    for (int r = 0; r < 8; ++r) {
        const int nrow = r + hl * 8;
        const size_t base = ((size_t)(tile * 16 + nrow) * AK + a) * COUT_;
        out[base + lm]      = acc0[r];
        out[base + 16 + lm] = acc1[r];
    }
}

extern "C" void kernel_launch(void* const* d_in, const int* in_sizes, int n_in,
                              void* d_out, int out_size, void* d_ws, size_t ws_size,
                              hipStream_t stream) {
    (void)in_sizes; (void)n_in; (void)d_ws; (void)ws_size; (void)out_size;
    const float* q_pts = (const float*)d_in[0];
    const float* s_pts = (const float*)d_in[1];
    const int*   nbi   = (const int*)  d_in[2];
    const float* x     = (const float*)d_in[3];
    const float* kp    = (const float*)d_in[4];
    const float* anc   = (const float*)d_in[5];
    const float* wts   = (const float*)d_in[6];
    float*       out   = (float*)d_out;

    dim3 grid(QN / 16);   // 1000 tiles of 16 queries
    dim3 block(128);      // 4 wave32 waves, one per anchor
    hipLaunchKernelGGL(kpconv_e2pn_fused, grid, block, SMEM_BYTES, stream,
                       q_pts, s_pts, nbi, x, kp, anc, wts, out);
}
